// RVAE_77455440216551
// MI455X (gfx1250) — compile-verified, matrix operand order checked by reading, not run on hardware
//
#include <hip/hip_runtime.h>
#include <cstdint>

#define Tn 512
#define Bn 256
#define Hn 128
#define G4 512
#define Xn 257
#define XPn 288
#define Zn 16

typedef __attribute__((ext_vector_type(16))) __bf16 v16bf;
typedef __attribute__((ext_vector_type(8)))  float  v8f;
typedef __attribute__((ext_vector_type(4)))  unsigned int v4u;
typedef __attribute__((ext_vector_type(8)))  int v8i;
typedef __attribute__((ext_vector_type(4)))  int v4i;

#if __has_builtin(__builtin_amdgcn_tensor_load_to_lds)
#define USE_TDM 1
#if __has_include(<hip/amd_detail/amd_gfx1250_TDM.h>)
#define TDM_SIX_ARGS 1
#endif
#endif

__device__ __forceinline__ unsigned short f2bf(float f) {
  unsigned int u = __float_as_uint(f);
  u += 0x7FFFu + ((u >> 16) & 1u);   // round-to-nearest-even
  return (unsigned short)(u >> 16);
}
__device__ __forceinline__ float sigm(float x) { return 1.0f / (1.0f + __expf(-x)); }
__device__ __forceinline__ float tanh_(float x) {
  float e = __expf(-2.0f * fabsf(x));
  float r = (1.0f - e) / (1.0f + e);
  return copysignf(r, x);
}
__device__ __forceinline__ v16bf ldb(const unsigned short* p) {
  return *reinterpret_cast<const v16bf*>(p);
}
__device__ __forceinline__ v8f wmma_bf16(v16bf a, v16bf b, v8f c) {
  return __builtin_amdgcn_wmma_f32_16x16x32_bf16(false, a, false, b, (short)0, c,
                                                 false, false);
}
__device__ __forceinline__ void cp_lds(unsigned short* dst, const unsigned short* src,
                                       int n_ush) {
  const uint4* s = reinterpret_cast<const uint4*>(src);
  uint4* d = reinterpret_cast<uint4*>(dst);
  const int n = n_ush >> 3;
  for (int i = threadIdx.x; i < n; i += blockDim.x) d[i] = s[i];
}

#ifdef USE_TDM
// 1-D contiguous TDM transfer: nbytes from gsrc -> LDS at lds_dst.
// D# per CDNA5 ISA ch.8: group0 {count=1, lds_addr, global_addr, type=2},
// group1 {data_size=2B, tensor_dim0=elems, tensor_dim1=1, tile_dim0=elems,
//         tensor_dim0_stride=elems}; groups 2/3 unused (<=2D).
__device__ __forceinline__ void tdm_load_1d(void* lds_dst, const void* gsrc,
                                            unsigned int nbytes) {
  const unsigned int lds_off = (unsigned int)(size_t)lds_dst;  // addr[31:0] = LDS offset
  const unsigned long long ga = (unsigned long long)(size_t)gsrc;
  const unsigned int elems = nbytes >> 1;  // 2-byte elements
  v4u g0 = {1u, lds_off, (unsigned int)ga,
            (unsigned int)(ga >> 32) | (2u << 30)};
  v8i g1 = {(int)(1u << 16),                          // data_size = 1 -> 2 bytes
            (int)((elems & 0xFFFFu) << 16),           // tensor_dim0[15:0]
            (int)((elems >> 16) | (1u << 16)),        // tensor_dim0[31:16] | tensor_dim1=1
            (int)((elems & 0xFFFFu) << 16),           // tile_dim0 (elems <= 65535)
            0,                                        // tile_dim1/2 unused
            (int)elems,                               // tensor_dim0_stride[31:0]
            0, 0};
  v4i gz = {0, 0, 0, 0};
#ifdef TDM_SIX_ARGS
  v8i gz8 = {0, 0, 0, 0, 0, 0, 0, 0};
  __builtin_amdgcn_tensor_load_to_lds(g0, g1, gz, gz, gz8, 0);
#else
  __builtin_amdgcn_tensor_load_to_lds(g0, g1, gz, gz, 0);
#endif
}
#endif

// ---------------------------------------------------------------------------
// Weight packing into per-lane-contiguous WMMA-B operand layout:
// dst[(ntile*ktiles + ktile)*512 + lane*16 + j],
//   n = ntile*16 + lane%16 ; k = ktile*32 + (lane/16)*16 + j
// ---------------------------------------------------------------------------
__global__ void pack_b_kernel(const float* __restrict__ W, int out_real, int in_real,
                              int ktiles, unsigned short* __restrict__ dst) {
  const int kt = blockIdx.x % ktiles;
  const int ntb = blockIdx.x / ktiles;
  const int base = blockIdx.x * 512;
#pragma unroll
  for (int s = 0; s < 2; ++s) {
    const int e = threadIdx.x * 2 + s;
    const int lane = e >> 4, j = e & 15;
    const int n = ntb * 16 + (lane & 15);
    const int k = kt * 32 + (lane >> 4) * 16 + j;
    float v = (n < out_real && k < in_real) ? W[(size_t)n * in_real + k] : 0.0f;
    dst[base + e] = f2bf(v);
  }
}

__global__ void addvec_kernel(const float* __restrict__ a, const float* __restrict__ b,
                              float* __restrict__ o) {
  const int i = blockIdx.x * 256 + threadIdx.x;
  o[i] = a[i] + b[i];
}

// x (B, X, T) f32  ->  xbf (T, B, XP) bf16, K padded with zeros
__global__ void xconv_kernel(const float* __restrict__ x,
                             unsigned short* __restrict__ xbf) {
  const size_t i = (size_t)blockIdx.x * 256 + threadIdx.x;
  const int k = (int)(i % XPn);
  const size_t r = i / XPn;
  const int b = (int)(r % Bn);
  const int t = (int)(r / Bn);
  float v = 0.0f;
  if (k < Xn) v = x[(size_t)b * (Xn * Tn) + (size_t)k * Tn + t];
  xbf[i] = f2bf(v);
}

__global__ void zero32_kernel(unsigned int* __restrict__ p) {
  p[(size_t)blockIdx.x * 256 + threadIdx.x] = 0u;
}

// ---------------------------------------------------------------------------
// LSTM recurrence. One WG = one 16-row batch tile, 8 waves, cell state in VGPRs.
// Input tile streamed into a ping-pong LDS buffer by the Tensor Data Mover one
// step ahead (pointer-swap double buffering: no per-step address recompute).
// Whh LDS-resident; Wih streamed from L2.  k-outer / 4-accumulator schedule.
// ---------------------------------------------------------------------------
template <int KT_IN>
__global__ void lstm_kernel(const unsigned short* __restrict__ inA,   // (T,B,KT_IN*32) bf16
                            const unsigned short* __restrict__ WihP,  // packed [32][KT_IN][512]
                            const unsigned short* __restrict__ WhhP,  // packed [32][4][512]
                            const float* __restrict__ bias,           // 512 = bih+bhh
                            unsigned short* __restrict__ hout,        // (T,B,128) bf16
                            int reverse) {
  constexpr int XUSH = 16 * KT_IN * 32;
  __shared__ alignas(32) unsigned short sX0[XUSH];           // ping
  __shared__ alignas(32) unsigned short sX1[XUSH];           // pong
  __shared__ alignas(32) unsigned short sWhh[32 * 4 * 512];  // 128 KB
  __shared__ alignas(32) unsigned short sH[16 * Hn];         // 4 KB
  __shared__ alignas(32) float sG[16 * G4];                  // 32 KB

  const int tid = threadIdx.x;
  const int lane = tid & 31;
  const int wave = tid >> 5;
  const int b0 = blockIdx.x * 16;
  const int row = lane & 15;
  const int khi = (lane >> 4) * 16;
  const int prow = tid & 15;
  const int u0 = (tid >> 4) * 8;

  cp_lds(sWhh, WhhP, 32 * 4 * 512);
  for (int i = tid; i < 16 * Hn; i += blockDim.x) sH[i] = 0;

  unsigned short* sXcur = sX0;
  unsigned short* sXnxt = sX1;

  const int t0 = reverse ? (Tn - 1) : 0;
#ifdef USE_TDM
  if (wave == 0) {
    tdm_load_1d(sXcur, inA + ((size_t)t0 * Bn + b0) * (KT_IN * 32), XUSH * 2);
    __builtin_amdgcn_s_wait_tensorcnt(0);
  }
#else
  cp_lds(sXcur, inA + ((size_t)t0 * Bn + b0) * (KT_IN * 32), XUSH);
#endif

  float c[8];
#pragma unroll
  for (int q = 0; q < 8; ++q) c[q] = 0.0f;
  __syncthreads();

  const unsigned short* wih_base = WihP + (size_t)(wave * 4 * KT_IN) * 512 + lane * 16;
  const unsigned short* whh_base = sWhh + (wave * 4 * 4) * 512 + lane * 16;

  for (int t = 0; t < Tn; ++t) {
    const int tt = reverse ? (Tn - 1 - t) : t;
    const int tnext = reverse ? (tt - 1) : (tt + 1);

#ifdef USE_TDM
    if (wave == 0 && t + 1 < Tn)  // async-prefetch next step's tile via TDM
      tdm_load_1d(sXnxt, inA + ((size_t)tnext * Bn + b0) * (KT_IN * 32), XUSH * 2);
#endif

    // gates(16x512): k-outer, 4 accumulators (one per n-tile of this wave)
    const unsigned short* sXrow = sXcur + row * (KT_IN * 32);
    v8f g0 = {}, g1 = {}, g2 = {}, g3 = {};
#pragma unroll
    for (int kk = 0; kk < KT_IN; ++kk) {
      v16bf a = ldb(sXrow + kk * 32 + khi);
      g0 = wmma_bf16(a, ldb(wih_base + (size_t)(0 * KT_IN + kk) * 512), g0);
      g1 = wmma_bf16(a, ldb(wih_base + (size_t)(1 * KT_IN + kk) * 512), g1);
      g2 = wmma_bf16(a, ldb(wih_base + (size_t)(2 * KT_IN + kk) * 512), g2);
      g3 = wmma_bf16(a, ldb(wih_base + (size_t)(3 * KT_IN + kk) * 512), g3);
    }
#pragma unroll
    for (int kk = 0; kk < 4; ++kk) {
      v16bf a = ldb(sH + row * Hn + kk * 32 + khi);
      g0 = wmma_bf16(a, ldb(whh_base + (0 * 4 + kk) * 512), g0);
      g1 = wmma_bf16(a, ldb(whh_base + (1 * 4 + kk) * 512), g1);
      g2 = wmma_bf16(a, ldb(whh_base + (2 * 4 + kk) * 512), g2);
      g3 = wmma_bf16(a, ldb(whh_base + (3 * 4 + kk) * 512), g3);
    }
    {
      const int rofs = (lane >> 4) * 8;
      const int colb = wave * 64 + row;
#pragma unroll
      for (int r = 0; r < 8; ++r) {
        sG[(r + rofs) * G4 + colb + 0]  = g0[r];
        sG[(r + rofs) * G4 + colb + 16] = g1[r];
        sG[(r + rofs) * G4 + colb + 32] = g2[r];
        sG[(r + rofs) * G4 + colb + 48] = g3[r];
      }
    }
    __syncthreads();

    // pointwise LSTM cell: thread owns (row=prow, units u0..u0+7)
#pragma unroll
    for (int q = 0; q < 8; ++q) {
      const int u = u0 + q;
      const float gi = sG[prow * G4 + u] + bias[u];
      const float gf = sG[prow * G4 + 128 + u] + bias[128 + u];
      const float gg = sG[prow * G4 + 256 + u] + bias[256 + u];
      const float go = sG[prow * G4 + 384 + u] + bias[384 + u];
      float cv = sigm(gf) * c[q] + sigm(gi) * tanh_(gg);
      c[q] = cv;
      const unsigned short hb = f2bf(sigm(go) * tanh_(cv));
      sH[prow * Hn + u] = hb;
      hout[((size_t)tt * Bn + b0 + prow) * Hn + u] = hb;
    }
#ifdef USE_TDM
    if (wave == 0) __builtin_amdgcn_s_wait_tensorcnt(0);  // next tile landed
#else
    if (t + 1 < Tn)
      cp_lds(sXnxt, inA + ((size_t)tnext * Bn + b0) * (KT_IN * 32), XUSH);
#endif
    __syncthreads();

    unsigned short* tmp = sXcur;  // ping-pong swap
    sXcur = sXnxt;
    sXnxt = tmp;
  }
}

// ---------------------------------------------------------------------------
// Inference scan: h=tanh([gx,gz]@W0^T+b0); mean/logvar heads; reparameterize;
// one g_z LSTM-cell step.  Weights LDS-resident; gx tile TDM ping-pong.
// ---------------------------------------------------------------------------
__global__ void inference_kernel(
    const unsigned short* __restrict__ gxh, const float* __restrict__ eps,
    const unsigned short* __restrict__ W0p, const unsigned short* __restrict__ Wmp,
    const unsigned short* __restrict__ Wvp, const unsigned short* __restrict__ WihZp,
    const unsigned short* __restrict__ WhhZp, const float* __restrict__ b0g,
    const float* __restrict__ bmg, const float* __restrict__ bvg,
    const float* __restrict__ bzg, float* __restrict__ out_mean,
    float* __restrict__ out_logvar, float* __restrict__ out_z,
    unsigned short* __restrict__ zbf) {
  constexpr int GUSH = 16 * Hn;
  __shared__ alignas(32) unsigned short sW0[8 * 8 * 512];    // 64 KB
  __shared__ alignas(32) unsigned short sWm[4 * 512];
  __shared__ alignas(32) unsigned short sWv[4 * 512];
  __shared__ alignas(32) unsigned short sWihZ[32 * 512];     // 32 KB
  __shared__ alignas(32) unsigned short sWhhZ[32 * 4 * 512]; // 128 KB
  __shared__ alignas(32) unsigned short sGX0[GUSH];          // gx ping
  __shared__ alignas(32) unsigned short sGX1[GUSH];          // gx pong
  __shared__ alignas(32) unsigned short sH[16 * Hn];
  __shared__ alignas(32) unsigned short sGZ[16 * Hn];
  __shared__ alignas(32) unsigned short sZ[16 * 32];
  __shared__ alignas(32) float sMV[2 * 16 * 16];
  __shared__ alignas(32) float sG[16 * G4];                  // 32 KB

  const int tid = threadIdx.x, lane = tid & 31, wave = tid >> 5;
  const int row = lane & 15, khi = (lane >> 4) * 16;
  const int b0 = blockIdx.x * 16;
  const int prow = tid & 15, u0 = (tid >> 4) * 8;

  cp_lds(sW0, W0p, 8 * 8 * 512);
  cp_lds(sWm, Wmp, 4 * 512);
  cp_lds(sWv, Wvp, 4 * 512);
  cp_lds(sWihZ, WihZp, 32 * 512);
  cp_lds(sWhhZ, WhhZp, 32 * 4 * 512);
  for (int i = tid; i < 16 * Hn; i += blockDim.x) sGZ[i] = 0;
  for (int i = tid; i < 16 * 32; i += blockDim.x) sZ[i] = 0;

  unsigned short* sGXcur = sGX0;
  unsigned short* sGXnxt = sGX1;
#ifdef USE_TDM
  if (wave == 0) {
    tdm_load_1d(sGXcur, gxh + (size_t)b0 * Hn, GUSH * 2);
    __builtin_amdgcn_s_wait_tensorcnt(0);
  }
#else
  cp_lds(sGXcur, gxh + (size_t)b0 * Hn, GUSH);
#endif

  float cz[8];
#pragma unroll
  for (int q = 0; q < 8; ++q) cz[q] = 0.0f;
  __syncthreads();

  for (int t = 0; t < Tn; ++t) {
#ifdef USE_TDM
    if (wave == 0 && t + 1 < Tn)
      tdm_load_1d(sGXnxt, gxh + ((size_t)(t + 1) * Bn + b0) * Hn, GUSH * 2);
#endif
    // phase 1: h = tanh(concat(gx_t, g_z) @ W0^T + b0); 8 waves x 1 n-tile
    {
      const unsigned short* grow = sGXcur + row * Hn;
      v8f acc = {};
#pragma unroll
      for (int kk = 0; kk < 4; ++kk) {
        v16bf a = ldb(grow + kk * 32 + khi);
        acc = wmma_bf16(a, ldb(sW0 + (wave * 8 + kk) * 512 + lane * 16), acc);
      }
#pragma unroll
      for (int kk = 0; kk < 4; ++kk) {
        v16bf a = ldb(sGZ + row * Hn + kk * 32 + khi);
        acc = wmma_bf16(a, ldb(sW0 + (wave * 8 + 4 + kk) * 512 + lane * 16), acc);
      }
      const int col = wave * 16 + row;
      const float bb = b0g[col];
      const int rofs = (lane >> 4) * 8;
#pragma unroll
      for (int r = 0; r < 8; ++r)
        sH[(r + rofs) * Hn + col] = f2bf(tanh_(acc[r] + bb));
    }
    __syncthreads();

    // phase 2: heads (wave0=mean, wave1=logvar); wave-uniform branch
    if (wave < 2) {
      const unsigned short* Wp = (wave == 0) ? sWm : sWv;
      v8f hacc = {};
#pragma unroll
      for (int kk = 0; kk < 4; ++kk) {
        v16bf a = ldb(sH + row * Hn + kk * 32 + khi);
        hacc = wmma_bf16(a, ldb(Wp + kk * 512 + lane * 16), hacc);
      }
      const int rofs = (lane >> 4) * 8;
#pragma unroll
      for (int r = 0; r < 8; ++r) sMV[wave * 256 + (r + rofs) * 16 + row] = hacc[r];
    }
    __syncthreads();

    // phase 3: reparameterize + emit mean/logvar/z
    {
      const int zd = tid >> 4;
      const float mval = sMV[prow * 16 + zd] + bmg[zd];
      const float lval = sMV[256 + prow * 16 + zd] + bvg[zd];
      const float e = eps[((size_t)t * Bn + b0 + prow) * Zn + zd];
      const float zval = e * __expf(0.5f * lval) + mval;
      const size_t ob = (size_t)(b0 + prow) * (Zn * Tn) + (size_t)zd * Tn + t;
      out_mean[ob] = mval;
      out_logvar[ob] = lval;
      out_z[ob] = zval;
      const unsigned short zb = f2bf(zval);
      sZ[prow * 32 + zd] = zb;
      zbf[((size_t)t * Bn + b0 + prow) * 32 + zd] = zb;
    }
    __syncthreads();

    // phase 4: rnn_g_z cell; k-outer, 4 accumulators
    {
      const unsigned short* wz_base = sWihZ + (wave * 4) * 512 + lane * 16;
      const unsigned short* wh_base = sWhhZ + (wave * 4 * 4) * 512 + lane * 16;
      v8f q0 = {}, q1 = {}, q2 = {}, q3 = {};
      {
        v16bf a = ldb(sZ + row * 32 + khi);
        q0 = wmma_bf16(a, ldb(wz_base + 0 * 512), q0);
        q1 = wmma_bf16(a, ldb(wz_base + 1 * 512), q1);
        q2 = wmma_bf16(a, ldb(wz_base + 2 * 512), q2);
        q3 = wmma_bf16(a, ldb(wz_base + 3 * 512), q3);
      }
#pragma unroll
      for (int kk = 0; kk < 4; ++kk) {
        v16bf a = ldb(sGZ + row * Hn + kk * 32 + khi);
        q0 = wmma_bf16(a, ldb(wh_base + (0 * 4 + kk) * 512), q0);
        q1 = wmma_bf16(a, ldb(wh_base + (1 * 4 + kk) * 512), q1);
        q2 = wmma_bf16(a, ldb(wh_base + (2 * 4 + kk) * 512), q2);
        q3 = wmma_bf16(a, ldb(wh_base + (3 * 4 + kk) * 512), q3);
      }
      const int rofs = (lane >> 4) * 8;
      const int colb = wave * 64 + row;
#pragma unroll
      for (int r = 0; r < 8; ++r) {
        sG[(r + rofs) * G4 + colb + 0]  = q0[r];
        sG[(r + rofs) * G4 + colb + 16] = q1[r];
        sG[(r + rofs) * G4 + colb + 32] = q2[r];
        sG[(r + rofs) * G4 + colb + 48] = q3[r];
      }
    }
    __syncthreads();
#pragma unroll
    for (int q = 0; q < 8; ++q) {
      const int u = u0 + q;
      const float gi = sG[prow * G4 + u] + bzg[u];
      const float gf = sG[prow * G4 + 128 + u] + bzg[128 + u];
      const float gg = sG[prow * G4 + 256 + u] + bzg[256 + u];
      const float go = sG[prow * G4 + 384 + u] + bzg[384 + u];
      float cv = sigm(gf) * cz[q] + sigm(gi) * tanh_(gg);
      cz[q] = cv;
      sGZ[prow * Hn + u] = f2bf(sigm(go) * tanh_(cv));
    }
#ifdef USE_TDM
    if (wave == 0) __builtin_amdgcn_s_wait_tensorcnt(0);
#endif
    __syncthreads();

    unsigned short* tmp = sGXcur;  // ping-pong swap
    sGXcur = sGXnxt;
    sGXnxt = tmp;
  }
}

// ---------------------------------------------------------------------------
// y = exp(hdec @ Wy^T + by): (T*B=131072) x 257 x 128 WMMA GEMM.
// 128-row A panel TDM'd into LDS; 8 waves sweep 8x17 tiles.
// ---------------------------------------------------------------------------
__global__ void ygemm_kernel(const unsigned short* __restrict__ hdec,
                             const unsigned short* __restrict__ WyP,
                             const float* __restrict__ by, float* __restrict__ y) {
  __shared__ alignas(32) unsigned short sA[128 * Hn];  // 32 KB
  const int tid = threadIdx.x, lane = tid & 31, wave = tid >> 5;
  const int row = lane & 15, khi = (lane >> 4) * 16;
  const size_t r0 = (size_t)blockIdx.x * 128;

#ifdef USE_TDM
  if (wave == 0) {
    tdm_load_1d(sA, hdec + r0 * Hn, 128 * Hn * 2);
    __builtin_amdgcn_s_wait_tensorcnt(0);
  }
#else
  cp_lds(sA, hdec + r0 * Hn, 128 * Hn);
#endif
  __syncthreads();

  for (int idx = wave; idx < 8 * 17; idx += 8) {
    const int mt = idx / 17, nt = idx % 17;
    v8f acc = {};
#pragma unroll
    for (int kk = 0; kk < 4; ++kk) {
      v16bf a = ldb(sA + (mt * 16 + row) * Hn + kk * 32 + khi);
      v16bf b = ldb(WyP + (size_t)(nt * 4 + kk) * 512 + lane * 16);
      acc = wmma_bf16(a, b, acc);
    }
    const int n = nt * 16 + row;
    if (n < Xn) {
      const float bn = by[n];
#pragma unroll
      for (int r = 0; r < 8; ++r) {
        const size_t R = r0 + mt * 16 + r + ((lane >> 4) * 8);  // row = t*B + b
        const int tt = (int)(R >> 8);
        const int bb = (int)(R & 255);
        y[(size_t)bb * ((size_t)Xn * Tn) + (size_t)n * Tn + tt] = __expf(acc[r] + bn);
      }
    }
  }
}

// ---------------------------------------------------------------------------
extern "C" void kernel_launch(void* const* d_in, const int* in_sizes, int n_in,
                              void* d_out, int out_size, void* d_ws, size_t ws_size,
                              hipStream_t stream) {
  (void)in_sizes; (void)n_in; (void)out_size; (void)ws_size;
  const float* x = (const float*)d_in[0];
  const float* eps = (const float*)d_in[1];
  const float* Wih_gx = (const float*)d_in[2];
  const float* Whh_gx = (const float*)d_in[3];
  const float* bih_gx = (const float*)d_in[4];
  const float* bhh_gx = (const float*)d_in[5];
  const float* Wih_gz = (const float*)d_in[6];
  const float* Whh_gz = (const float*)d_in[7];
  const float* bih_gz = (const float*)d_in[8];
  const float* bhh_gz = (const float*)d_in[9];
  const float* W0 = (const float*)d_in[10];
  const float* b0 = (const float*)d_in[11];
  const float* Wm = (const float*)d_in[12];
  const float* bm = (const float*)d_in[13];
  const float* Wv = (const float*)d_in[14];
  const float* bv = (const float*)d_in[15];
  const float* Wih_h = (const float*)d_in[16];
  const float* Whh_h = (const float*)d_in[17];
  const float* bih_h = (const float*)d_in[18];
  const float* bhh_h = (const float*)d_in[19];
  const float* Wy = (const float*)d_in[20];
  const float* by = (const float*)d_in[21];

  char* ws = (char*)d_ws;
  size_t off = 0;
  auto alloc = [&](size_t bytes) -> char* {
    char* r = ws + off;
    off += (bytes + 255) & ~(size_t)255;
    return r;
  };
  unsigned short* WhhGXp = (unsigned short*)alloc(32 * 4 * 512 * 2);
  unsigned short* WihGXp = (unsigned short*)alloc(32 * 9 * 512 * 2);
  unsigned short* WhhGZp = (unsigned short*)alloc(32 * 4 * 512 * 2);
  unsigned short* WihGZp = (unsigned short*)alloc(32 * 1 * 512 * 2);
  unsigned short* WhhHp  = (unsigned short*)alloc(32 * 4 * 512 * 2);
  unsigned short* WihHp  = (unsigned short*)alloc(32 * 1 * 512 * 2);
  unsigned short* W0p    = (unsigned short*)alloc(8 * 8 * 512 * 2);
  unsigned short* Wmp    = (unsigned short*)alloc(1 * 4 * 512 * 2);
  unsigned short* Wvp    = (unsigned short*)alloc(1 * 4 * 512 * 2);
  unsigned short* Wyp    = (unsigned short*)alloc(17 * 4 * 512 * 2);
  float* bGX = (float*)alloc(512 * 4);
  float* bGZ = (float*)alloc(512 * 4);
  float* bH  = (float*)alloc(512 * 4);
  unsigned short* xbfp  = (unsigned short*)alloc((size_t)Tn * Bn * XPn * 2);
  unsigned short* gxhp  = (unsigned short*)alloc((size_t)Tn * Bn * Hn * 2);
  unsigned short* zbfp  = (unsigned short*)alloc((size_t)Tn * Bn * 32 * 2);
  unsigned short* hdecp = (unsigned short*)alloc((size_t)Tn * Bn * Hn * 2);

  float* yo = (float*)d_out;
  float* mo = yo + (size_t)Bn * Xn * Tn;
  float* lo = mo + (size_t)Bn * Zn * Tn;
  float* zo = lo + (size_t)Bn * Zn * Tn;

  // prep: pack weights (bf16, WMMA-B layout), combine biases, convert x
  pack_b_kernel<<<32 * 4, 256, 0, stream>>>(Whh_gx, 512, 128, 4, WhhGXp);
  pack_b_kernel<<<32 * 9, 256, 0, stream>>>(Wih_gx, 512, 257, 9, WihGXp);
  pack_b_kernel<<<32 * 4, 256, 0, stream>>>(Whh_gz, 512, 128, 4, WhhGZp);
  pack_b_kernel<<<32 * 1, 256, 0, stream>>>(Wih_gz, 512, 16, 1, WihGZp);
  pack_b_kernel<<<32 * 4, 256, 0, stream>>>(Whh_h, 512, 128, 4, WhhHp);
  pack_b_kernel<<<32 * 1, 256, 0, stream>>>(Wih_h, 512, 16, 1, WihHp);
  pack_b_kernel<<<8 * 8, 256, 0, stream>>>(W0, 128, 256, 8, W0p);
  pack_b_kernel<<<1 * 4, 256, 0, stream>>>(Wm, 16, 128, 4, Wmp);
  pack_b_kernel<<<1 * 4, 256, 0, stream>>>(Wv, 16, 128, 4, Wvp);
  pack_b_kernel<<<17 * 4, 256, 0, stream>>>(Wy, 257, 128, 4, Wyp);
  addvec_kernel<<<2, 256, 0, stream>>>(bih_gx, bhh_gx, bGX);
  addvec_kernel<<<2, 256, 0, stream>>>(bih_gz, bhh_gz, bGZ);
  addvec_kernel<<<2, 256, 0, stream>>>(bih_h, bhh_h, bH);
  xconv_kernel<<<(Tn * Bn * XPn) / 256, 256, 0, stream>>>(x, xbfp);
  zero32_kernel<<<((Tn * Bn * 32) / 2) / 256, 256, 0, stream>>>((unsigned int*)zbfp);

  // phase 1: backward LSTM over x
  lstm_kernel<9><<<Bn / 16, 256, 0, stream>>>(xbfp, WihGXp, WhhGXp, bGX, gxhp, 1);
  // phase 2: inference scan
  inference_kernel<<<Bn / 16, 256, 0, stream>>>(gxhp, eps, W0p, Wmp, Wvp, WihGZp,
                                                WhhGZp, b0, bm, bv, bGZ, mo, lo, zo,
                                                zbfp);
  // phase 3: forward decoder LSTM over z
  lstm_kernel<1><<<Bn / 16, 256, 0, stream>>>(zbfp, WihHp, WhhHp, bH, hdecp, 0);
  // phase 4: y = exp(hdec @ Wy^T + by)
  ygemm_kernel<<<(Tn * Bn) / 128, 256, 0, stream>>>(hdecp, Wyp, by, yo);
}
